// MoE_36146444763329
// MI455X (gfx1250) — matrix operand order repaired and checked by reading, not compiled, not run on hardware
//
#include <hip/hip_runtime.h>

#define OMEGA0f 30.0f
#define BB 2
#define NN 16384
#define EE 8
#define HH 256
#define NL 3            // hidden layers (L-1)
#define MT 128          // point rows per block
#define TPB 256         // 8 wave32s

typedef __attribute__((ext_vector_type(16))) _Float16 v16h;
typedef __attribute__((ext_vector_type(8)))  _Float16 v8h;
typedef __attribute__((ext_vector_type(4)))  _Float16 v4h;
typedef __attribute__((ext_vector_type(8)))  float    v8f;
typedef __attribute__((ext_vector_type(4)))  unsigned int u32x4;
typedef __attribute__((ext_vector_type(8)))  int          i32x8;
typedef __attribute__((ext_vector_type(4)))  int          i32x4;

union AFrag { v16h v; v8h h[2]; };

#if __has_builtin(__builtin_amdgcn_tensor_load_to_lds) && __has_builtin(__builtin_amdgcn_s_wait_tensorcnt)
#define HAVE_TDM 1
#else
#define HAVE_TDM 0
#endif

// LDS layout: weights f16 [HH][HH] | act ping [MT][HH] f16 | act pong | xs[MT] f32 | bias[HH] f32
#define LDS_W_BYTES   (HH * HH * 2)
#define LDS_ACT_BYTES (MT * HH * 2)
#define LDS_BYTES     (LDS_W_BYTES + 2 * LDS_ACT_BYTES + MT * 4 + HH * 4)

#if HAVE_TDM
// DMA one 256x256 f16 tile (global, row-major) -> LDS[lds_off], via Tensor Data Mover.
__device__ __forceinline__ void tdm_load_w16(unsigned lds_off, unsigned long long gaddr)
{
    // D# group 0: count=1 | lds_addr | global_addr[56:0] | type=2
    u32x4 g0 = { 1u,
                 lds_off,
                 (unsigned)(gaddr & 0xFFFFFFFFull),
                 (unsigned)(((gaddr >> 32) & 0x1FFFFFFull) | (2u << 30)) };
    // D# group 1: data_size=2B; tensor_dim0=256; tensor_dim1=256; tile_dim0=256;
    //             tile_dim1=256; tensor_dim0_stride=256 (elements)
    i32x8 g1 = { (int)(1u << 16),        // [17:16] data_size = 2 bytes
                 (int)(256u << 16),      // [63:48] tensor_dim0[15:0]
                 (int)(256u << 16),      // [95:80] tensor_dim1[15:0]
                 (int)(256u << 16),      // [127:112] tile_dim0
                 (int)256,               // [143:128] tile_dim1 (tile_dim2=0)
                 (int)256,               // [191:160] tensor_dim0_stride[31:0]
                 0, 0 };
    i32x4 z4 = { 0, 0, 0, 0 };
#if defined(__clang_major__) && __clang_major__ >= 23
    i32x8 z8 = { 0, 0, 0, 0, 0, 0, 0, 0 };
    __builtin_amdgcn_tensor_load_to_lds(g0, g1, z4, z4, z8, 0);
#else
    __builtin_amdgcn_tensor_load_to_lds(g0, g1, z4, z4, 0);
#endif
    __builtin_amdgcn_s_wait_tensorcnt(0);
}
#endif

__global__ __launch_bounds__(TPB, 1)
void moe_siren_kernel(const float* __restrict__ x,
                      const float* __restrict__ gate_w,
                      const float* __restrict__ gate_b,
                      const float* __restrict__ w0,
                      const float* __restrict__ b0,
                      const float* __restrict__ wh,
                      const float* __restrict__ bh,
                      const float* __restrict__ wo,
                      const float* __restrict__ bo,
                      const _Float16* __restrict__ wh16,   // pre-converted weights (or null)
                      float* __restrict__ dst,
                      int use_ws16,
                      int use_atomic)
{
    extern __shared__ __align__(16) char smem[];
    _Float16* wlds  = (_Float16*)smem;                       // [HH][HH] row-major (out,in)
    _Float16* act0  = (_Float16*)(smem + LDS_W_BYTES);       // [MT][HH]
    _Float16* act1  = act0 + MT * HH;
    float*    xs    = (float*)(smem + LDS_W_BYTES + 2 * LDS_ACT_BYTES);
    float*    biasl = xs + MT;                               // [HH]

    const int tid    = threadIdx.x;
    const int tilesN = NN / MT;
    const int tile   = blockIdx.x % tilesN;
    const int e      = (blockIdx.x / tilesN) % EE;
    const int b      = blockIdx.x / (tilesN * EE);
    const int n0     = tile * MT;
    const float* xp  = x + (size_t)b * NN + n0;

    if (tid < MT) xs[tid] = xp[tid];
    __syncthreads();

    // ---- first sine layer: act0[r][tid] = sin(w0*x + b0), thread owns column `tid`
    {
        const float w0v = w0[(size_t)e * HH + tid];
        const float b0v = b0[(size_t)e * HH + tid];
        #pragma unroll 4
        for (int r = 0; r < MT; ++r)
            act0[r * HH + tid] = (_Float16)__sinf(OMEGA0f * (xs[r] * w0v + b0v));
    }

    const int lane    = tid & 31;
    const int ln      = lane & 15;     // row (A) / column (B) within 16-tile
    const int hi      = lane >> 4;     // half-wave select
    const int rowbase = (tid >> 5) * 16;

    _Float16* bufs[2] = { act0, act1 };

    for (int l = 0; l < NL; ++l) {
        const _Float16* cur = bufs[l & 1];
        _Float16*       nxt = bufs[(l + 1) & 1];

        __syncthreads();   // prior layer done reading wlds/biasl, writing cur
        biasl[tid] = bh[((size_t)l * EE + e) * HH + tid];
        if (use_ws16) {
            const _Float16* W16 = wh16 + ((size_t)l * EE + e) * HH * HH;
#if HAVE_TDM
            // one TDM descriptor per block fills the 128KB weight buffer; scalar
            // branch (readfirstlane) so exactly one wave issues it (TDM ignores EXEC)
            if (__builtin_amdgcn_readfirstlane(tid) < 32)
                tdm_load_w16((unsigned)(uintptr_t)wlds,
                             (unsigned long long)(uintptr_t)W16);
#else
            for (int idx = tid * 8; idx < HH * HH; idx += TPB * 8)
                *(v8h*)(wlds + idx) = *(const v8h*)(W16 + idx);
#endif
        } else {
            const float* W = wh + ((size_t)l * EE + e) * HH * HH;
            for (int idx = tid * 4; idx < HH * HH; idx += TPB * 4) {
                const float4 v = *(const float4*)(W + idx);
                v4h h4 = { (_Float16)v.x, (_Float16)v.y, (_Float16)v.z, (_Float16)v.w };
                *(v4h*)(wlds + idx) = h4;
            }
            if (l + 1 < NL) {   // hint next layer's weights toward L2
                const float* Wn = wh + ((size_t)(l + 1) * EE + e) * HH * HH;
                #pragma unroll
                for (int q = 0; q < 4; ++q)
                    __builtin_prefetch(Wn + (size_t)tid * 256 + q * 64, 0, 1);
            }
        }
        __syncthreads();

        // A fragments: this wave's 16 rows, full K=256, contiguous k order (same as B)
        // lanes 0-15 hold K {kb+0..15}; lanes 16-31 hold {kb+16..31}
        AFrag a[8];
        #pragma unroll
        for (int kt = 0; kt < 8; ++kt) {
            const _Float16* arow = cur + (rowbase + ln) * HH + kt * 32 + hi * 16;
            a[kt].h[0] = *(const v8h*)(arow);
            a[kt].h[1] = *(const v8h*)(arow + 8);
        }

        #pragma unroll 2
        for (int ct = 0; ct < 16; ++ct) {
            // preload ALL B fragments for this output tile so the 16 ds_load_b128
            // stay in flight while the WMMA chain drains them (partial dscnt waits)
            const _Float16* brow = wlds + (ct * 16 + ln) * HH + hi * 16;
            v16h bf[8];
            #pragma unroll
            for (int kt = 0; kt < 8; ++kt)
                bf[kt] = *(const v16h*)(brow + kt * 32);

            v8f c = {};
            #pragma unroll
            for (int kt = 0; kt < 8; ++kt)
                c = __builtin_amdgcn_wmma_f32_16x16x32_f16(
                        false, a[kt].v, false, bf[kt], (short)0, c, false, false);

            const int   col = ct * 16 + ln;
            const float bv  = biasl[col];
            float sv[8];
            #pragma unroll
            for (int r = 0; r < 8; ++r)
                sv[r] = __sinf(OMEGA0f * (c[r] + bv));
            #pragma unroll
            for (int r = 0; r < 8; ++r) {
                const int m = hi ? (r + 8) : r;   // C layout: lanes 16-31 hold rows 8..15
                nxt[(rowbase + m) * HH + col] = (_Float16)sv[r];
            }
        }
    }
    __syncthreads();

    // ---- epilogue: final linear (H->1), gate softmax, per-expert partial
    if (tid < MT) {
        const _Float16* hfin = bufs[NL & 1] + tid * HH;
        const float*    wov  = wo + (size_t)e * HH;
        float acc = bo[e];
        #pragma unroll 4
        for (int j = 0; j < HH; j += 8) {
            const v8h hv = *(const v8h*)(hfin + j);
            #pragma unroll
            for (int k = 0; k < 8; ++k) acc += (float)hv[k] * wov[j + k];
        }
        const float xv = xs[tid];
        float lg[EE], mx = -1e30f;
        #pragma unroll
        for (int je = 0; je < EE; ++je) {
            lg[je] = xv * gate_w[je] + gate_b[je];
            mx = fmaxf(mx, lg[je]);
        }
        float sum = 0.f, we = 0.f;
        #pragma unroll
        for (int je = 0; je < EE; ++je) {
            const float p = __expf(lg[je] - mx);
            sum += p;
            if (je == e) we = p;
        }
        const float contrib = (we / sum) * acc;
        const int   n       = n0 + tid;
        if (use_atomic) atomicAdd(dst + (size_t)b * NN + n, contrib);
        else            dst[((size_t)b * EE + e) * NN + n] = contrib;
    }
}

// one-time (per launch) f32 -> f16 weight conversion into workspace
__global__ void convert_wh_kernel(const float* __restrict__ wh,
                                  _Float16* __restrict__ wh16)
{
    const int i = (blockIdx.x * blockDim.x + threadIdx.x) * 4;
    if (i >= NL * EE * HH * HH) return;
    const float4 v = *(const float4*)(wh + i);
    v4h h4 = { (_Float16)v.x, (_Float16)v.y, (_Float16)v.z, (_Float16)v.w };
    *(v4h*)(wh16 + i) = h4;
}

__global__ void reduce_experts_kernel(const float* __restrict__ part,
                                      float* __restrict__ out)
{
    const int i = blockIdx.x * blockDim.x + threadIdx.x;
    if (i >= BB * NN) return;
    const int b = i / NN, n = i % NN;
    float s = 0.f;
    #pragma unroll
    for (int e = 0; e < EE; ++e) s += part[((size_t)b * EE + e) * NN + n];
    out[i] = s;
}

__global__ void zero_out_kernel(float* __restrict__ out)
{
    const int i = blockIdx.x * blockDim.x + threadIdx.x;
    if (i < BB * NN) out[i] = 0.f;
}

extern "C" void kernel_launch(void* const* d_in, const int* in_sizes, int n_in,
                              void* d_out, int out_size, void* d_ws, size_t ws_size,
                              hipStream_t stream)
{
    (void)in_sizes; (void)n_in; (void)out_size;
    const float* x      = (const float*)d_in[0];
    const float* gate_w = (const float*)d_in[1];
    const float* gate_b = (const float*)d_in[2];
    const float* w0     = (const float*)d_in[3];
    const float* b0     = (const float*)d_in[4];
    const float* wh     = (const float*)d_in[5];
    const float* bh     = (const float*)d_in[6];
    const float* wo     = (const float*)d_in[7];
    const float* bo     = (const float*)d_in[8];
    float*       out    = (float*)d_out;

    (void)hipFuncSetAttribute((const void*)moe_siren_kernel,
                              hipFuncAttributeMaxDynamicSharedMemorySize, LDS_BYTES);

    const dim3   grid(BB * EE * (NN / MT));
    const size_t wh16_bytes = (size_t)NL * EE * HH * HH * sizeof(_Float16); // 3 MB
    const size_t part_bytes = (size_t)BB * EE * NN * sizeof(float);         // 1 MB
    const int    nconv      = (NL * EE * HH * HH) / 4;

    if (ws_size >= wh16_bytes + part_bytes) {
        _Float16* wh16 = (_Float16*)d_ws;
        float*    part = (float*)((char*)d_ws + wh16_bytes);
        convert_wh_kernel<<<(nconv + 255) / 256, 256, 0, stream>>>(wh, wh16);
        moe_siren_kernel<<<grid, TPB, LDS_BYTES, stream>>>(
            x, gate_w, gate_b, w0, b0, wh, bh, wo, bo, wh16, part, 1, 0);
        reduce_experts_kernel<<<(BB * NN + 255) / 256, 256, 0, stream>>>(part, out);
    } else if (ws_size >= part_bytes) {
        float* part = (float*)d_ws;
        moe_siren_kernel<<<grid, TPB, LDS_BYTES, stream>>>(
            x, gate_w, gate_b, w0, b0, wh, bh, wo, bo, (const _Float16*)nullptr,
            part, 0, 0);
        reduce_experts_kernel<<<(BB * NN + 255) / 256, 256, 0, stream>>>(part, out);
    } else {
        zero_out_kernel<<<(BB * NN + 255) / 256, 256, 0, stream>>>(out);
        moe_siren_kernel<<<grid, TPB, LDS_BYTES, stream>>>(
            x, gate_w, gate_b, w0, b0, wh, bh, wo, bo, (const _Float16*)nullptr,
            out, 0, 1);
    }
}
